// Attention_171798692097
// MI455X (gfx1250) — matrix operand order repaired and checked by reading, not compiled, not run on hardware
//
#include <hip/hip_runtime.h>

// Monotonic-window multi-head attention for MI455X (gfx1250, wave32, WMMA + TDM).
// Outputs (concatenated in d_out, fp32):
//   R          [B,Tq,2D]     offset 0
//   alignments [B,H,Tq,Tk]   offset ALN_OFF   (single NT-store pass)
//   max_attn   [B,H,Tq]      offset AMAX_OFF  (indices stored as float)

typedef __attribute__((ext_vector_type(16))) __bf16 v16bf;
typedef __attribute__((ext_vector_type(8)))  float  v8f;
typedef __attribute__((ext_vector_type(4)))  float  v4f;
typedef int v4i __attribute__((vector_size(16)));             // matches builtin pointee
typedef __attribute__((address_space(1))) v4i* gv4p;          // global int4*
typedef __attribute__((address_space(3))) v4i* lv4p;          // LDS int4*
typedef unsigned int u32x4 __attribute__((vector_size(16)));  // TDM D# group types
typedef int i32x4 __attribute__((vector_size(16)));
typedef int i32x8 __attribute__((vector_size(32)));

#define HEADS 8
#define WIN   3
#define BB    32
#define TQ    512
#define TK    512
#define DD    512
#define DH    64
#define QT    16               // q rows per wave (one WMMA M tile)
#define NWAVE 4                // waves per block
#define QBLK  (QT*NWAVE)       // 64 q rows per block
#define STA_STRIDE 20          // fp32 stage row stride (80B: 16B-aligned, bank-skewed)
#define STB_STRIDE 40          // bf16 stage row stride (80B)

#if __has_builtin(__builtin_amdgcn_tensor_load_to_lds)
#define USE_TDM 1
#else
#define USE_TDM 0
#endif

#if __has_builtin(__builtin_amdgcn_global_load_async_to_lds_b128)
#define USE_ASYNC_LDS 1
#else
#define USE_ASYNC_LDS 0
#endif

#if __has_builtin(__builtin_amdgcn_s_wait_asynccnt)
#define WAIT_ASYNC() __builtin_amdgcn_s_wait_asynccnt(0)
#else
#define WAIT_ASYNC() asm volatile("s_wait_asynccnt 0" ::: "memory")
#endif

#if __has_builtin(__builtin_amdgcn_s_wait_tensorcnt)
#define WAIT_TENSOR() __builtin_amdgcn_s_wait_tensorcnt(0)
#else
#define WAIT_TENSOR() asm volatile("s_wait_tensorcnt 0" ::: "memory")
#endif

union FragU { unsigned short u[16]; unsigned int w[8]; v16bf v; };

#define WMMA_BF16(A, B, C) \
  __builtin_amdgcn_wmma_f32_16x16x32_bf16(false, (A).v, false, (B).v, (short)0, (C), false, false)

__device__ __forceinline__ unsigned short f2bf(float f) {
  union { float f; unsigned u; } c; c.f = f;
  return (unsigned short)(c.u >> 16);
}

#if USE_TDM
// One TDM descriptor moves the whole strided fp32 head slice (tile 64 x 512,
// tensor_dim0_stride = 512 elements) into LDS, tracked on TENSORcnt.
// D# packing per CDNA5 ISA section 8.3 (group0) / 8.4 (group1); groups 2,3 zero (2-D).
__device__ __forceinline__ void tdm_load_head(const void* gsrc, unsigned lds_off) {
  const unsigned long long ga = (unsigned long long)gsrc;
  const u32x4 g0 = { 1u,                                    // count=1 (valid), user mode
                     lds_off,                               // lds_addr (bytes)
                     (unsigned)(ga & 0xFFFFFFFFu),          // global_addr[31:0]
                     (unsigned)((ga >> 32) & 0x01FFFFFFu)   // global_addr[56:32]
                         | (2u << 30) };                    // type=2 ("image")
  const i32x8 g1 = { (int)(2u << 16),          // wg_mask=0, data_size=2 (4 bytes)
                     (int)((unsigned)DH << 16),// tensor_dim0 = 64  (bits [79:48], low16)
                     (int)((unsigned)TK << 16),// tensor_dim1 = 512 (bits [111:80], low16)
                     (int)((unsigned)DH << 16),// tile_dim0 = 64    (bits [127:112])
                     (int)TK,                  // tile_dim1 = 512   (bits [143:128])
                     (int)DD,                  // tensor_dim0_stride = 512 (bits [207:160])
                     0, 0 };
  const i32x4 g2 = { 0, 0, 0, 0 };
  const i32x4 g3 = { 0, 0, 0, 0 };
#if defined(__clang_major__) && (__clang_major__ >= 23)
  const i32x8 g4 = { 0, 0, 0, 0, 0, 0, 0, 0 };
  __builtin_amdgcn_tensor_load_to_lds(g0, g1, g2, g3, g4, 0);
#else
  __builtin_amdgcn_tensor_load_to_lds(g0, g1, g2, g3, 0);
#endif
}
#endif

// 16 contiguous bf16 (32B, 16B-aligned) -> two ds_load_b128
__device__ __forceinline__ void ld_frag(FragU& f, const unsigned short* p) {
  const uint4 a = *(const uint4*)p;
  const uint4 b = *(const uint4*)(p + 8);
  f.w[0] = a.x; f.w[1] = a.y; f.w[2] = a.z; f.w[3] = a.w;
  f.w[4] = b.x; f.w[5] = b.y; f.w[6] = b.z; f.w[7] = b.w;
}

// two separate 16B chunks (A-fragment K-split layout)
__device__ __forceinline__ void ld_frag2(FragU& f, const unsigned short* p0,
                                         const unsigned short* p1) {
  const uint4 a = *(const uint4*)p0;
  const uint4 b = *(const uint4*)p1;
  f.w[0] = a.x; f.w[1] = a.y; f.w[2] = a.z; f.w[3] = a.w;
  f.w[4] = b.x; f.w[5] = b.y; f.w[6] = b.z; f.w[7] = b.w;
}

// S-tile[16x16] = Qtile[16x64] x K^T[64x16]  (two K=32 bf16 WMMAs, f32 accum)
__device__ __forceinline__ v8f score_tile(const FragU* qa, const unsigned short* Klds,
                                          int kt, int lane) {
  const int key = kt * 16 + (lane & 15);
  const int g16 = (lane >> 4) << 4;
  v8f acc = {};
  FragU kb;
  ld_frag2(kb, &Klds[key * DH + (g16 >> 1)], &Klds[key * DH + 16 + (g16 >> 1)]);
  acc = WMMA_BF16(qa[0], kb, acc);
  ld_frag2(kb, &Klds[key * DH + 32 + (g16 >> 1)], &Klds[key * DH + 48 + (g16 >> 1)]);
  acc = WMMA_BF16(qa[1], kb, acc);
  return acc;
}

// fp32 row-major [TK][stride] -> bf16 [TK][DH] (vectorized b128 read / b64 write)
__device__ __forceinline__ void convertK(const float* src, int stride,
                                         unsigned short* Klds, int tid) {
  for (int c = tid * 4; c < TK * DH; c += 512) {
    const int key = c >> 6, d = c & 63;
    const float4 f = *(const float4*)(src + (size_t)key * stride + d);
    ushort4 o; o.x = f2bf(f.x); o.y = f2bf(f.y); o.z = f2bf(f.z); o.w = f2bf(f.w);
    *(ushort4*)&Klds[key * DH + d] = o;
  }
}

// fp32 row-major [TK][stride] -> bf16 transposed [DH][TK]
__device__ __forceinline__ void convertVt(const float* src, int stride,
                                          unsigned short* Vt, int tid) {
  for (int c = tid * 4; c < TK * DH; c += 512) {
    const int key = c >> 6, d = c & 63;
    const float4 f = *(const float4*)(src + (size_t)key * stride + d);
    Vt[(d + 0) * TK + key] = f2bf(f.x);
    Vt[(d + 1) * TK + key] = f2bf(f.y);
    Vt[(d + 2) * TK + key] = f2bf(f.z);
    Vt[(d + 3) * TK + key] = f2bf(f.w);
  }
}

__global__ void __launch_bounds__(128)
monotonic_attn_kernel(const float* __restrict__ Q, const float* __restrict__ K,
                      const float* __restrict__ V, const int* __restrict__ PM,
                      float* __restrict__ out)
{
  extern __shared__ char smem_raw[];
  unsigned short* Klds = (unsigned short*)smem_raw;                       // 64KB [TK][DH]
  unsigned short* Vt   = (unsigned short*)(smem_raw + 65536);             // 64KB [DH][TK]
  float*          stA0 = (float*)(smem_raw + 131072);                     // 5KB  fp32 stage
  unsigned short* stB0 = (unsigned short*)(smem_raw + 131072 + 5120);     // 5KB  bf16 stage
  float*          Fst  = (float*)(smem_raw + 131072 + 10240);             // 128KB DMA stage

  const size_t ALN_OFF  = (size_t)BB * TQ * 2 * DD;
  const size_t AMAX_OFF = ALN_OFF + (size_t)BB * HEADS * TQ * TK;

  const int tid  = threadIdx.x;
  const int lane = tid & 31;
  const int wv   = tid >> 5;
  const int qb   = blockIdx.x & 7;
  const int h    = (blockIdx.x >> 3) & 7;
  const int b    = blockIdx.x >> 6;

  float*          stA = stA0 + wv * 16 * STA_STRIDE;
  unsigned short* stB = stB0 + wv * 16 * STB_STRIDE;

  int nvalid = PM[h * BB + b] + WIN + 1;        // column-only monotonic window
  if (nvalid > TK) nvalid = TK;
  const int nvt = (nvalid + 15) >> 4;

  const float* Kh = K + (size_t)b * TK * DD + h * DH;
  const float* Vh = V + (size_t)b * TK * DD + h * DH;

#if USE_TDM
  {
    // Tensor Data Mover: one descriptor per head slice, fp32 -> LDS stage,
    // then VALU converts LDS->LDS bf16 (+ transpose for V).
    const unsigned lds_off = (unsigned)(unsigned long long)Fst;  // low32 flat = LDS offset
    if (wv == 0) {
      tdm_load_head(Kh, lds_off);
      WAIT_TENSOR();
    }
    __syncthreads();
    convertK(Fst, DH, Klds, tid);
    __syncthreads();
    if (wv == 0) {
      tdm_load_head(Vh, lds_off);
      WAIT_TENSOR();
    }
    __syncthreads();
    convertVt(Fst, DH, Vt, tid);
    __syncthreads();
  }
#elif USE_ASYNC_LDS
  {
    // Fallback: per-lane async DMA loads (ASYNCcnt path).
    lv4p lbase = (lv4p)(unsigned int)(unsigned long long)Fst;
    auto issue = [&](const float* src) {
      gv4p g = (gv4p)(unsigned long long)src;
      for (int c = tid; c < (TK * DH) / 4; c += 128) {   // 8192 x 16B chunks
        const int key = c >> 4, part = c & 15;           // head row, 16B chunk in row
        __builtin_amdgcn_global_load_async_to_lds_b128(
            g + key * (DD / 4) + part, lbase + c, 0, 0);
      }
    };
    issue(Kh);
    WAIT_ASYNC();
    __syncthreads();
    convertK(Fst, DH, Klds, tid);
    __syncthreads();
    issue(Vh);
    WAIT_ASYNC();
    __syncthreads();
    convertVt(Fst, DH, Vt, tid);
    __syncthreads();
  }
#else
  (void)Fst;
  convertK(Kh, DD, Klds, tid);
  convertVt(Vh, DD, Vt, tid);
  __syncthreads();
#endif

  const int q0  = qb * QBLK + wv * QT;
  const int row = lane & 15;
  const int grp = lane >> 4;

  // ---- Q A-fragments (16x32 x2 over dh=64), pre-scaled by 1/sqrt(dh)=0.125 ----
  // A bf16 layout: lane<16 -> K {0..7,16..23}; lane>=16 -> K {8..15,24..31}
  FragU qa[2];
  {
    const float* Qr = Q + (size_t)(b * TQ + q0 + row) * DD + h * DH;
#pragma unroll
    for (int f = 0; f < 2; ++f) {
      const float4 a0 = *(const float4*)(Qr + f * 32 + grp * 8);
      const float4 a1 = *(const float4*)(Qr + f * 32 + grp * 8 + 4);
      const float4 b0 = *(const float4*)(Qr + f * 32 + 16 + grp * 8);
      const float4 b1 = *(const float4*)(Qr + f * 32 + 16 + grp * 8 + 4);
      qa[f].u[0]  = f2bf(a0.x * 0.125f); qa[f].u[1]  = f2bf(a0.y * 0.125f);
      qa[f].u[2]  = f2bf(a0.z * 0.125f); qa[f].u[3]  = f2bf(a0.w * 0.125f);
      qa[f].u[4]  = f2bf(a1.x * 0.125f); qa[f].u[5]  = f2bf(a1.y * 0.125f);
      qa[f].u[6]  = f2bf(a1.z * 0.125f); qa[f].u[7]  = f2bf(a1.w * 0.125f);
      qa[f].u[8]  = f2bf(b0.x * 0.125f); qa[f].u[9]  = f2bf(b0.y * 0.125f);
      qa[f].u[10] = f2bf(b0.z * 0.125f); qa[f].u[11] = f2bf(b0.w * 0.125f);
      qa[f].u[12] = f2bf(b1.x * 0.125f); qa[f].u[13] = f2bf(b1.y * 0.125f);
      qa[f].u[14] = f2bf(b1.z * 0.125f); qa[f].u[15] = f2bf(b1.w * 0.125f);
    }
  }

  // ---- Pass A: online softmax stats, lane-local via LDS transpose ----
  // Lane l owns row (l&15); its half of the columns is (l>>4)*8 .. +7.
  float m1 = -1e30f, l1 = 0.f;
  int   idx1 = 0;
  const int cb = grp * 8;

  for (int kt = 0; kt < nvt; ++kt) {
    v8f s = score_tile(qa, Klds, kt, lane);
    // C-layout -> row-major fp32 stage (ds_store_2addr pairs)
#pragma unroll
    for (int r = 0; r < 8; ++r)
      stA[(r + 8 * grp) * STA_STRIDE + (lane & 15)] = s[r];
    asm volatile("s_wait_dscnt 0" ::: "memory");
    // row-major gather: 8 contiguous floats = 2x ds_load_b128
    const float4 a0 = *(const float4*)&stA[row * STA_STRIDE + cb];
    const float4 a1 = *(const float4*)&stA[row * STA_STRIDE + cb + 4];
    float vv[8] = {a0.x, a0.y, a0.z, a0.w, a1.x, a1.y, a1.z, a1.w};
    const int colb = kt * 16 + cb;
    float tm = -1e30f; int ti = 0;
#pragma unroll
    for (int j = 0; j < 8; ++j) {
      const float x = (colb + j < nvalid) ? vv[j] : -1e30f;
      vv[j] = x;
      const bool g = x > tm;               // strict > keeps first index
      ti = g ? (colb + j) : ti;
      tm = g ? x : tm;
    }
    // merge the two half-rows (lanes l <-> l+16): branchless
    {
      const float om = __shfl_xor(tm, 16, 32);
      const int   oi = __shfl_xor(ti, 16, 32);
      const bool take = (om > tm) || ((om == tm) && (oi < ti));
      tm = take ? om : tm;
      ti = take ? oi : ti;
    }
    const float mnew = fmaxf(m1, tm);
    float es = 0.f;
#pragma unroll
    for (int j = 0; j < 8; ++j) es += __expf(vv[j] - mnew);
    es += __shfl_xor(es, 16, 32);
    l1   = l1 * __expf(m1 - mnew) + es;
    idx1 = (tm > m1) ? ti : idx1;          // ties keep earlier (first occurrence)
    m1   = mnew;
  }
  const float rl1 = 1.0f / l1;

  // argmax output: lane i < 16 owns row q0+i
  if (lane < 16)
    __builtin_nontemporal_store(
        (float)idx1, &out[AMAX_OFF + (size_t)((b * HEADS + h) * TQ + q0 + lane)]);

  // broadcast per-row stats into C-layout per-lane arrays for pass B
  float mB[8], rlB[8];
#pragma unroll
  for (int r = 0; r < 8; ++r) {
    const int src = r + 8 * grp;
    mB[r]  = __shfl(m1, src, 32);
    rlB[r] = __shfl(rl1, src, 32);
  }

  // ---- Pass B: recompute S, emit normalized P once (NT), accumulate O = P x V ----
  v8f oacc[4] = {};
  for (int kt = 0; kt < TK / 16; ++kt) {
    float p[8];
    if (kt < nvt) {
      v8f s = score_tile(qa, Klds, kt, lane);
      const int col = kt * 16 + (lane & 15);
#pragma unroll
      for (int r = 0; r < 8; ++r)
        p[r] = (col < nvalid) ? __expf(s[r] - mB[r]) * rlB[r] : 0.f;
    } else {
#pragma unroll
      for (int r = 0; r < 8; ++r) p[r] = 0.f;   // masked region: exact zeros
    }
#pragma unroll
    for (int r = 0; r < 8; ++r) {
      const int qr = q0 + r + 8 * grp;
      __builtin_nontemporal_store(
          p[r], &out[ALN_OFF + (size_t)((b * HEADS + h) * TQ + qr) * TK + kt * 16 + (lane & 15)]);
      stB[(r + 8 * grp) * STB_STRIDE + (kt & 1) * 16 + (lane & 15)] = f2bf(p[r]);
    }
    if (kt & 1) {
      asm volatile("s_wait_dscnt 0" ::: "memory");
      if ((kt - 1) * 16 < nvalid) {              // uniform per wave: skip all-zero pairs
        FragU pa;
        ld_frag2(pa, &stB[(lane & 15) * STB_STRIDE + grp * 8],
                     &stB[(lane & 15) * STB_STRIDE + 16 + grp * 8]);
        const int kbase = (kt - 1) * 16;
#pragma unroll
        for (int nt = 0; nt < 4; ++nt) {
          FragU vb;
          ld_frag2(vb, &Vt[(nt * 16 + (lane & 15)) * TK + kbase + (grp << 3)],
                       &Vt[(nt * 16 + (lane & 15)) * TK + kbase + 16 + (grp << 3)]);
          oacc[nt] = WMMA_BF16(pa, vb, oacc[nt]);
        }
      }
    }
  }

  // ---- heads_out slice of R (streamed, NT) ----
#pragma unroll
  for (int nt = 0; nt < 4; ++nt)
#pragma unroll
    for (int r = 0; r < 8; ++r)
      __builtin_nontemporal_store(
          oacc[nt][r],
          &out[(size_t)(b * TQ + q0 + r + 8 * grp) * (2 * DD) + h * DH + nt * 16 + (lane & 15)]);

  // ---- R second half = Q copy (h==0 blocks own it), b128 NT stores ----
  if (h == 0) {
    const int qbase = qb * QBLK;
    const float* Qb = Q + (size_t)(b * TQ + qbase) * DD;
    float* Rb = out + (size_t)(b * TQ + qbase) * (2 * DD) + DD;
    for (int e = tid * 4; e < QBLK * DD; e += 512) {
      const int qr = e >> 9, d = e & 511;
      const v4f f = *(const v4f*)(Qb + (size_t)qr * DD + d);
      __builtin_nontemporal_store(f, (v4f*)(Rb + (size_t)qr * (2 * DD) + d));
    }
  }
}

extern "C" void kernel_launch(void* const* d_in, const int* in_sizes, int n_in,
                              void* d_out, int out_size, void* d_ws, size_t ws_size,
                              hipStream_t stream) {
  (void)in_sizes; (void)n_in; (void)out_size; (void)d_ws; (void)ws_size;
  const float* Q  = (const float*)d_in[0];
  const float* K  = (const float*)d_in[1];
  const float* V  = (const float*)d_in[2];
  const int*   PM = (const int*)d_in[3];     // JAX x64 disabled -> int32 [H,B]
  float* out = (float*)d_out;

  const dim3 grid(BB * HEADS * (TQ / QBLK));  // 2048 blocks
  const dim3 block(128);                      // 4 waves (wave32)
  // 64K K + 64K Vt + 5K stA + 5K stB + 128K DMA fp32 stage = 266KB (< 320KB/WGP)
  const size_t shmem = 131072 + 5120 + 5120 + 131072;
  monotonic_attn_kernel<<<grid, block, shmem, stream>>>(Q, K, V, PM, out);
}
